// PerTokenFFN_42743514530356
// MI455X (gfx1250) — matrix operand order repaired and checked by reading, not compiled, not run on hardware
//
#include <hip/hip_runtime.h>
#include <math.h>

// Problem constants (match the reference)
#define BATCH 256
#define TOK   64
#define DDIM  512
#define HDIM  2048

// CDNA5 WMMA types
typedef __attribute__((ext_vector_type(16))) __bf16 v16bf;
typedef __attribute__((ext_vector_type(2)))  __bf16 v2bf;
typedef __attribute__((ext_vector_type(8)))  float  v8f;

// TDM descriptor vector types (per CDNA5_HIP.md probe results)
typedef __attribute__((ext_vector_type(4))) unsigned int u32x4;
typedef __attribute__((ext_vector_type(8))) int          i32x8;
typedef __attribute__((ext_vector_type(4))) int          i32x4;

#ifndef __has_builtin
#define __has_builtin(x) 0
#endif

#if __has_builtin(__builtin_amdgcn_tensor_load_to_lds) && \
    __has_builtin(__builtin_amdgcn_s_wait_tensorcnt)
#define HAVE_TDM 1
#else
#define HAVE_TDM 0
#endif

// therock-10.0 headers ship the TDM header and the 6-arg builtin;
// ROCm 7.2 has the 5-arg builtin (per compile probes).
#if __has_include(<hip/amd_detail/amd_gfx1250_TDM.h>)
#define TDM_ARGS6 1
#else
#define TDM_ARGS6 0
#endif

union Frag16 {
    v16bf v;
    uint4 q[2];
};

// native packed f32 -> bf16 (RNE); lowers to v_cvt_pk_bf16_f32 on gfx1250
__device__ __forceinline__ unsigned int pk2bf(float a, float b) {
    v2bf r;
    r[0] = (__bf16)a;
    r[1] = (__bf16)b;
    return __builtin_bit_cast(unsigned int, r);
}

__device__ __forceinline__ v8f zero8() {
    v8f z;
#pragma unroll
    for (int i = 0; i < 8; ++i) z[i] = 0.0f;
    return z;
}

#define LDK 40  // 32 K elements + 8 pad (ushorts) per LDS row (80 B pitch)

#if HAVE_TDM
// Issue a TDM load of a 128-row x 32-element bf16 tile into LDS.
// Tensor: 2D, element = 2 bytes, row pitch = TOK*HDIM elements.
// LDS padding: every 16 DWORDs (one 32-elem row) insert 4 DWORDs -> LDK=40 pitch.
__device__ __forceinline__ void tdm_load_A(unsigned ldsAddr, const unsigned short* g) {
    unsigned long long ga = (unsigned long long)(uintptr_t)g;
    u32x4 g0;
    g0[0] = 1u;                                       // count=1, user descriptor
    g0[1] = ldsAddr;                                  // LDS byte address
    g0[2] = (unsigned)ga;                             // global_addr[31:0]
    g0[3] = (unsigned)((ga >> 32) & 0x01FFFFFFull)    // global_addr[56:32]
          | (2u << 30);                               // type=2 ("image")
    i32x8 g1;
    g1[0] = (1 << 16)      // data_size = 2 bytes
          | (1 << 20)      // pad_enable
          | (3 << 22)      // pad_interval: 16 DWORDs
          | (3 << 25);     // pad_amount:   4 DWORDs
    g1[1] = (int)(((unsigned)HDIM & 0xFFFFu) << 16);            // tensor_dim0 lo
    g1[2] = (int)(((unsigned)HDIM >> 16) |
                  (((unsigned)(BATCH * TOK) & 0xFFFFu) << 16)); // td0 hi | td1 lo
    g1[3] = (int)(((unsigned)(BATCH * TOK) >> 16) | (32u << 16)); // td1 hi | tile_dim0=32
    g1[4] = 128;                                                // tile_dim1=128, tile_dim2=0
    g1[5] = (int)(TOK * HDIM);                                  // tensor_dim0_stride lo
    g1[6] = 0;
    g1[7] = 0;
    i32x4 z4 = {0, 0, 0, 0};
#if TDM_ARGS6
    i32x8 z8 = {0, 0, 0, 0, 0, 0, 0, 0};
    __builtin_amdgcn_tensor_load_to_lds(g0, g1, z4, z4, z8, 0);
#else
    __builtin_amdgcn_tensor_load_to_lds(g0, g1, z4, z4, 0);
#endif
}
#endif  // HAVE_TDM

// ---------------------------------------------------------------------------
// GEMM1 + bias + exact GELU:  hidden[b,t,h] = gelu(x[b,t,:] @ fc1[t,:,h] + b1)
// Block tile 128(M=batch) x 128(N=H), K-step 32 over D=512, double-buffered.
// Operands are f32 in HBM -> convert to bf16 in flight (packed cvt) while
// staging into LDS; B is stored [n][k] so each lane's WMMA-B fragment is two
// 16-byte LDS reads from one [n] row.
// ---------------------------------------------------------------------------
__global__ __launch_bounds__(256)
void ffn_gemm1_gelu(const float* __restrict__ x,
                    const float* __restrict__ fc1,
                    const float* __restrict__ bias1,
                    unsigned short* __restrict__ hidden)
{
    __shared__ __align__(16) unsigned short sA[2][128][LDK]; // [m][k]
    __shared__ __align__(16) unsigned short sB[2][128][LDK]; // [n][k]

    const int NT = HDIM / 128;   // 16
    const int MT = BATCH / 128;  // 2
    const int bid  = blockIdx.x;
    const int t    = bid / (NT * MT);
    const int rem  = bid % (NT * MT);
    const int mBlk = (rem / NT) * 128;
    const int nBlk = (rem % NT) * 128;

    const int tid  = threadIdx.x;
    const int lane = tid & 31;
    const int w    = tid >> 5;        // wave 0..7
    const int wm   = (w & 1) * 64;    // wave M offset within block tile
    const int wn   = (w >> 1) * 32;   // wave N offset within block tile
    const int sel  = lane >> 4;       // half-wave selector (ISA fragment layout)
    const int l15  = lane & 15;

    v8f acc[4][2];
#pragma unroll
    for (int i = 0; i < 4; ++i)
#pragma unroll
        for (int j = 0; j < 2; ++j) acc[i][j] = zero8();

    const float* gA = x   + (size_t)t * DDIM;          // + row*(T*D) + d
    const float* gB = fc1 + (size_t)t * DDIM * HDIM;   // + d*H + h

    const int NSTEP = DDIM / 32;  // 16

    // ---- prologue: stage K-step 0 into LDS buffer 0 ----
#pragma unroll
    for (int p = 0; p < 4; ++p) {
        const int r = (tid >> 3) + 32 * p;   // 0..127
        const int c = (tid & 7) * 4;         // 0..28
        float4 v4 = *(const float4*)(gA + (size_t)(mBlk + r) * (TOK * DDIM) + c);
        uint2 pa;
        pa.x = pk2bf(v4.x, v4.y);
        pa.y = pk2bf(v4.z, v4.w);
        *(uint2*)&sA[0][r][c] = pa;
    }
#pragma unroll
    for (int p = 0; p < 2; ++p) {
        const int kp = (tid >> 5) + 8 * p;   // k pair 0..15 -> k = 2kp, 2kp+1
        const int h  = (tid & 31) * 4;       // 0..124 (N)
        float4 r0 = *(const float4*)(gB + (size_t)(2 * kp)     * HDIM + nBlk + h);
        float4 r1 = *(const float4*)(gB + (size_t)(2 * kp + 1) * HDIM + nBlk + h);
        *(unsigned*)&sB[0][h + 0][2 * kp] = pk2bf(r0.x, r1.x);
        *(unsigned*)&sB[0][h + 1][2 * kp] = pk2bf(r0.y, r1.y);
        *(unsigned*)&sB[0][h + 2][2 * kp] = pk2bf(r0.z, r1.z);
        *(unsigned*)&sB[0][h + 3][2 * kp] = pk2bf(r0.w, r1.w);
    }
    __syncthreads();

    float4 aReg[4], bReg[2][2];

    for (int s = 0; s < NSTEP; ++s) {
        const int buf = s & 1;

        if (s + 1 < NSTEP) {
            const int d0 = (s + 1) * 32;
#pragma unroll
            for (int p = 0; p < 4; ++p) {
                const int r = (tid >> 3) + 32 * p;
                const int c = (tid & 7) * 4;
                aReg[p] = *(const float4*)(gA + (size_t)(mBlk + r) * (TOK * DDIM) + d0 + c);
            }
#pragma unroll
            for (int p = 0; p < 2; ++p) {
                const int kp = (tid >> 5) + 8 * p;
                const int h  = (tid & 31) * 4;
                bReg[p][0] = *(const float4*)(gB + (size_t)(d0 + 2 * kp)     * HDIM + nBlk + h);
                bReg[p][1] = *(const float4*)(gB + (size_t)(d0 + 2 * kp + 1) * HDIM + nBlk + h);
            }
            if (s + 4 < NSTEP) // pull future weight tile toward GL2
                __builtin_prefetch(gB + (size_t)((s + 4) * 32 + (tid >> 5)) * HDIM + nBlk, 0, 1);
        }

        // ---- compute: 4x2 WMMA fragments per wave ----
        Frag16 afr[4], bfr[2];
#pragma unroll
        for (int mf = 0; mf < 4; ++mf) {
            const unsigned short* base = &sA[buf][wm + mf * 16 + l15][0];
            afr[mf].q[0] = *(const uint4*)(base + sel * 8);        // K {0..7}/{8..15}
            afr[mf].q[1] = *(const uint4*)(base + 16 + sel * 8);   // K {16..23}/{24..31}
        }
#pragma unroll
        for (int nf = 0; nf < 2; ++nf) {
            const unsigned short* base = &sB[buf][wn + nf * 16 + l15][0];
            bfr[nf].q[0] = *(const uint4*)(base + sel * 8);        // K {0..7}/{8..15}
            bfr[nf].q[1] = *(const uint4*)(base + 16 + sel * 8);   // K {16..23}/{24..31}
        }
#pragma unroll
        for (int mf = 0; mf < 4; ++mf)
#pragma unroll
            for (int nf = 0; nf < 2; ++nf)
                acc[mf][nf] = __builtin_amdgcn_wmma_f32_16x16x32_bf16(
                    false, afr[mf].v, false, bfr[nf].v,
                    (short)0, acc[mf][nf], false, false);

        if (s + 1 < NSTEP) {
            const int nb = (s + 1) & 1;
#pragma unroll
            for (int p = 0; p < 4; ++p) {
                const int r = (tid >> 3) + 32 * p;
                const int c = (tid & 7) * 4;
                uint2 pa;
                pa.x = pk2bf(aReg[p].x, aReg[p].y);
                pa.y = pk2bf(aReg[p].z, aReg[p].w);
                *(uint2*)&sA[nb][r][c] = pa;
            }
#pragma unroll
            for (int p = 0; p < 2; ++p) {
                const int kp = (tid >> 5) + 8 * p;
                const int h  = (tid & 31) * 4;
                *(unsigned*)&sB[nb][h + 0][2 * kp] = pk2bf(bReg[p][0].x, bReg[p][1].x);
                *(unsigned*)&sB[nb][h + 1][2 * kp] = pk2bf(bReg[p][0].y, bReg[p][1].y);
                *(unsigned*)&sB[nb][h + 2][2 * kp] = pk2bf(bReg[p][0].z, bReg[p][1].z);
                *(unsigned*)&sB[nb][h + 3][2 * kp] = pk2bf(bReg[p][0].w, bReg[p][1].w);
            }
            __syncthreads();
        }
    }

    // ---- epilogue: bias + exact-erf GELU, store bf16 hidden ----
#pragma unroll
    for (int mf = 0; mf < 4; ++mf) {
#pragma unroll
        for (int nf = 0; nf < 2; ++nf) {
            const int col = nBlk + wn + nf * 16 + l15;
            const float bz = bias1[t * HDIM + col];
#pragma unroll
            for (int i = 0; i < 8; ++i) {
                const int row = mBlk + wm + mf * 16 + sel * 8 + i;
                const float hval = acc[mf][nf][i] + bz;
                const float g = 0.5f * hval * (1.0f + erff(hval * 0.70710678118654752f));
                hidden[(size_t)(row * TOK + t) * HDIM + col] =
                    (unsigned short)(pk2bf(g, 0.0f) & 0xFFFFu);
            }
        }
    }
}

// ---------------------------------------------------------------------------
// GEMM2 + bias:  out[b,t,d] = hidden[b,t,:] @ fc2[t,:,d] + b2
// A (hidden) is already bf16 -> staged into LDS by the Tensor Data Mover
// (TENSOR_LOAD_TO_LDS issued by wave 0, synchronized with s_wait_tensorcnt),
// with the D# pad fields reproducing the LDK=40 padded LDS pitch.
// B (fc2) is f32 -> packed-converted while staging, as in GEMM1.
// ---------------------------------------------------------------------------
__global__ __launch_bounds__(256)
void ffn_gemm2(const unsigned short* __restrict__ hidden,
               const float* __restrict__ fc2,
               const float* __restrict__ bias2,
               float* __restrict__ out)
{
    __shared__ __align__(16) unsigned short sA[2][128][LDK]; // [m][k]
    __shared__ __align__(16) unsigned short sB[2][128][LDK]; // [n][k]

    const int NT = DDIM / 128;   // 4
    const int MT = BATCH / 128;  // 2
    const int bid  = blockIdx.x;
    const int t    = bid / (NT * MT);
    const int rem  = bid % (NT * MT);
    const int mBlk = (rem / NT) * 128;
    const int nBlk = (rem % NT) * 128;

    const int tid  = threadIdx.x;
    const int lane = tid & 31;
    const int w    = tid >> 5;
    const int wm   = (w & 1) * 64;
    const int wn   = (w >> 1) * 32;
    const int sel  = lane >> 4;
    const int l15  = lane & 15;

    v8f acc[4][2];
#pragma unroll
    for (int i = 0; i < 4; ++i)
#pragma unroll
        for (int j = 0; j < 2; ++j) acc[i][j] = zero8();

    const unsigned short* gA = hidden + (size_t)t * HDIM;        // + row*(T*H) + h
    const float*          gB = fc2    + (size_t)t * HDIM * DDIM; // + h*D + d
    const unsigned short* gAblk = gA + (size_t)mBlk * (TOK * HDIM);

    const int NSTEP = HDIM / 32;  // 64

    // ---- prologue: K-step 0 ----
#if HAVE_TDM
    if (w == 0)
        tdm_load_A((unsigned)(uintptr_t)&sA[0][0][0], gAblk);
#else
#pragma unroll
    for (int p = 0; p < 2; ++p) {
        const int r = (tid >> 2) + 64 * p;   // 0..127
        const int c = (tid & 3) * 8;         // 0..24 (bf16 units)
        uint4 v4 = *(const uint4*)(gAblk + (size_t)r * (TOK * HDIM) + c);
        *(uint4*)&sA[0][r][c] = v4;
    }
#endif
#pragma unroll
    for (int p = 0; p < 2; ++p) {
        const int kp = (tid >> 5) + 8 * p;   // k pair 0..15
        const int n  = (tid & 31) * 4;       // 0..124 (N = d)
        float4 r0 = *(const float4*)(gB + (size_t)(2 * kp)     * DDIM + nBlk + n);
        float4 r1 = *(const float4*)(gB + (size_t)(2 * kp + 1) * DDIM + nBlk + n);
        *(unsigned*)&sB[0][n + 0][2 * kp] = pk2bf(r0.x, r1.x);
        *(unsigned*)&sB[0][n + 1][2 * kp] = pk2bf(r0.y, r1.y);
        *(unsigned*)&sB[0][n + 2][2 * kp] = pk2bf(r0.z, r1.z);
        *(unsigned*)&sB[0][n + 3][2 * kp] = pk2bf(r0.w, r1.w);
    }
#if HAVE_TDM
    if (w == 0) __builtin_amdgcn_s_wait_tensorcnt(0);
#endif
    __syncthreads();

#if !HAVE_TDM
    uint4 aReg[2];
#endif
    float4 bReg[2][2];

    for (int s = 0; s < NSTEP; ++s) {
        const int buf = s & 1;

        if (s + 1 < NSTEP) {
            const int k0 = (s + 1) * 32;
            const int nb = (s + 1) & 1;
#if HAVE_TDM
            if (w == 0)
                tdm_load_A((unsigned)(uintptr_t)&sA[nb][0][0], gAblk + k0);
#else
#pragma unroll
            for (int p = 0; p < 2; ++p) {
                const int r = (tid >> 2) + 64 * p;
                const int c = (tid & 3) * 8;
                aReg[p] = *(const uint4*)(gAblk + (size_t)r * (TOK * HDIM) + k0 + c);
            }
#endif
#pragma unroll
            for (int p = 0; p < 2; ++p) {
                const int kp = (tid >> 5) + 8 * p;
                const int n  = (tid & 31) * 4;
                bReg[p][0] = *(const float4*)(gB + (size_t)(k0 + 2 * kp)     * DDIM + nBlk + n);
                bReg[p][1] = *(const float4*)(gB + (size_t)(k0 + 2 * kp + 1) * DDIM + nBlk + n);
            }
            if (s + 4 < NSTEP)
                __builtin_prefetch(gB + (size_t)((s + 4) * 32 + (tid >> 5)) * DDIM + nBlk, 0, 1);
        }

        Frag16 afr[4], bfr[2];
#pragma unroll
        for (int mf = 0; mf < 4; ++mf) {
            const unsigned short* base = &sA[buf][wm + mf * 16 + l15][0];
            afr[mf].q[0] = *(const uint4*)(base + sel * 8);
            afr[mf].q[1] = *(const uint4*)(base + 16 + sel * 8);
        }
#pragma unroll
        for (int nf = 0; nf < 2; ++nf) {
            const unsigned short* base = &sB[buf][wn + nf * 16 + l15][0];
            bfr[nf].q[0] = *(const uint4*)(base + sel * 8);
            bfr[nf].q[1] = *(const uint4*)(base + 16 + sel * 8);
        }
#pragma unroll
        for (int mf = 0; mf < 4; ++mf)
#pragma unroll
            for (int nf = 0; nf < 2; ++nf)
                acc[mf][nf] = __builtin_amdgcn_wmma_f32_16x16x32_bf16(
                    false, afr[mf].v, false, bfr[nf].v,
                    (short)0, acc[mf][nf], false, false);

        if (s + 1 < NSTEP) {
            const int nb = (s + 1) & 1;
#if !HAVE_TDM
#pragma unroll
            for (int p = 0; p < 2; ++p) {
                const int r = (tid >> 2) + 64 * p;
                const int c = (tid & 3) * 8;
                *(uint4*)&sA[nb][r][c] = aReg[p];
            }
#endif
#pragma unroll
            for (int p = 0; p < 2; ++p) {
                const int kp = (tid >> 5) + 8 * p;
                const int n  = (tid & 31) * 4;
                *(unsigned*)&sB[nb][n + 0][2 * kp] = pk2bf(bReg[p][0].x, bReg[p][1].x);
                *(unsigned*)&sB[nb][n + 1][2 * kp] = pk2bf(bReg[p][0].y, bReg[p][1].y);
                *(unsigned*)&sB[nb][n + 2][2 * kp] = pk2bf(bReg[p][0].z, bReg[p][1].z);
                *(unsigned*)&sB[nb][n + 3][2 * kp] = pk2bf(bReg[p][0].w, bReg[p][1].w);
            }
#if HAVE_TDM
            if (w == 0) __builtin_amdgcn_s_wait_tensorcnt(0);
#endif
            __syncthreads();
        }
    }

    // ---- epilogue: bias2, f32 store ----
#pragma unroll
    for (int mf = 0; mf < 4; ++mf) {
#pragma unroll
        for (int nf = 0; nf < 2; ++nf) {
            const int col = nBlk + wn + nf * 16 + l15;
            const float bz = bias2[t * DDIM + col];
#pragma unroll
            for (int i = 0; i < 8; ++i) {
                const int row = mBlk + wm + mf * 16 + sel * 8 + i;
                out[(size_t)(row * TOK + t) * DDIM + col] = acc[mf][nf][i] + bz;
            }
        }
    }
}

extern "C" void kernel_launch(void* const* d_in, const int* in_sizes, int n_in,
                              void* d_out, int out_size, void* d_ws, size_t ws_size,
                              hipStream_t stream) {
    const float* x     = (const float*)d_in[0];
    const float* fc1   = (const float*)d_in[1];
    const float* bias1 = (const float*)d_in[2];
    const float* fc2   = (const float*)d_in[3];
    const float* bias2 = (const float*)d_in[4];
    float* out = (float*)d_out;

    // hidden activations in bf16: B*T*H*2 = 64 MB in workspace
    unsigned short* hidden = (unsigned short*)d_ws;

    dim3 blk(256);
    ffn_gemm1_gelu<<<dim3(TOK * (BATCH / 128) * (HDIM / 128)), blk, 0, stream>>>(
        x, fc1, bias1, hidden);
    ffn_gemm2<<<dim3(TOK * (BATCH / 128) * (DDIM / 128)), blk, 0, stream>>>(
        hidden, fc2, bias2, out);
}
